// TAB_Attention_AFS_67903432949954
// MI455X (gfx1250) — matrix operand order repaired and checked by reading, not compiled, not run on hardware
//
#include <hip/hip_runtime.h>
#include <hip/hip_bf16.h>

// ---------------------------------------------------------------------------
// AFT-Simple attention for MI455X (gfx1250, wave32, WMMA).
//   x  : [N=64, C=128, L=4096] f32
//   Wq/Wk/Wv : [128,128] f32, bq/bk/bv : [128] f32
//   out[n,d,l] = sigmoid(q[n,d,l]) * pooled[n,d]
//   where qT = Wq^T @ x[n]  (so output needs no transpose),
//   pooled[n,d] = sum_l exp(kT[d,l]) * vT[d,l] / sum_l exp(kT[d,l]).
// GEMMs in bf16 via V_WMMA_F32_16X16X32_BF16; everything else f32.
// B-fragments are double-buffered (unroll-by-2 ping-pong) so ds_load latency
// hides under the WMMA + exp epilogue instead of s_wait_dscnt 0 stalls.
// ---------------------------------------------------------------------------

typedef __attribute__((ext_vector_type(16))) __bf16 v16bf;
typedef __attribute__((ext_vector_type(8)))  float  v8f;

#define DIM   128
#define SEQ   4096
#define NB    64
#define LT    256            // L-chunk per workgroup
#define LTILES (LT / 16)     // 16 WMMA column-tiles per chunk (even -> unroll by 2)
#define LDST  136            // padded LDS row stride in bf16 elems (16B-aligned, bank-spread)

union FragB16 { v16bf v; uint4 q[2]; };

// A-matrix 16x32 bf16 fragment (per ISA layout):
// lanes 0-15 hold row M=lane, K = {0..7, 16..23}; lanes 16-31 row M=lane-16, K = {8..15, 24..31}.
__device__ __forceinline__ v16bf load_fragA(const __bf16* p) {
    FragB16 f;
    f.q[0] = *(const uint4*)(p);
    f.q[1] = *(const uint4*)(p + 16);
    return f.v;
}

// B-matrix 32x16 bf16 fragment: lane holds column N=lane&15, K = {0..7, 16..23} + (half?8:0),
// read from the K (=c) dimension of xT[l][c].
__device__ __forceinline__ v16bf load_fragB(const __bf16* p) {
    FragB16 f;
    f.q[0] = *(const uint4*)(p);
    f.q[1] = *(const uint4*)(p + 16);
    return f.v;
}

// Load all 4 k-step B fragments for one 16-column l-tile.
__device__ __forceinline__ void load_fragB4(v16bf b[4], const __bf16* bp) {
#pragma unroll
    for (int ks = 0; ks < 4; ++ks) b[ks] = load_fragB(bp + ks * 32);
}

__device__ __forceinline__ v8f wmma_bf16(v16bf a, v16bf b, v8f c) {
    // 8 args: (neg_a, A, neg_b, B, c_mod, C, reuse_a, reuse_b)
    return __builtin_amdgcn_wmma_f32_16x16x32_bf16(false, a, false, b, (short)0, c,
                                                   false, false);
}

// Stage W^T (bf16) into LDS: Wt[d*LDST + c] = W[c*DIM + d]
__device__ __forceinline__ void stage_weightT(const float* __restrict__ W,
                                              __bf16* Wt, int tid) {
    for (int i = tid; i < DIM * DIM; i += 256) {
        int c = i >> 7;
        int d = i & (DIM - 1);
        Wt[d * LDST + c] = (__bf16)W[i];      // global read coalesced over d
    }
}

// Stage one x tile (f32 -> bf16, transposed): xT[l*LDST + c] = x[n, c, l0 + l]
__device__ __forceinline__ void stage_xT(const float* __restrict__ xblk,
                                         __bf16* xT, int tid) {
    for (int i = tid; i < DIM * (LT / 4); i += 256) {
        int c = i >> 6;              // 0..127
        int g = (i & 63) << 2;       // l within chunk, step 4
        const float4 f = *(const float4*)(xblk + (size_t)c * SEQ + g);
        xT[(g + 0) * LDST + c] = (__bf16)f.x;
        xT[(g + 1) * LDST + c] = (__bf16)f.y;
        xT[(g + 2) * LDST + c] = (__bf16)f.z;
        xT[(g + 3) * LDST + c] = (__bf16)f.w;
    }
}

// ---------------------------------------------------------------------------
// Kernel 1: zero the pooled num/den workspace.
// ---------------------------------------------------------------------------
__global__ void aft_zero_kernel(float* __restrict__ p, int count) {
    int i = blockIdx.x * blockDim.x + threadIdx.x;
    if (i < count) p[i] = 0.0f;
}

// ---------------------------------------------------------------------------
// Kernel 2: K/V projections + online exp-pool.  grid = (SEQ/LT, NB), 256 thr.
// ---------------------------------------------------------------------------
__global__ __launch_bounds__(256) void aft_pool_kernel(
    const float* __restrict__ x,
    const float* __restrict__ Wk, const float* __restrict__ bk,
    const float* __restrict__ Wv, const float* __restrict__ bv,
    float* __restrict__ wsNum, float* __restrict__ wsDen)
{
    extern __shared__ __align__(16) char smem[];
    __bf16* xT  = (__bf16*)(smem);                                   // LT  x LDST
    __bf16* WtK = (__bf16*)(smem + (size_t)LT * LDST * 2);           // DIM x LDST
    __bf16* WtV = (__bf16*)(smem + (size_t)(LT + DIM) * LDST * 2);   // DIM x LDST

    const int tid = threadIdx.x;
    const int n   = blockIdx.y;
    const int l0  = blockIdx.x * LT;

    stage_weightT(Wk, WtK, tid);
    stage_weightT(Wv, WtV, tid);
    stage_xT(x + (size_t)n * DIM * SEQ + l0, xT, tid);
    __syncthreads();

    const int wave  = tid >> 5;          // 8 waves, one 16-row d-strip each
    const int lane  = tid & 31;
    const int half  = lane >> 4;
    const int l16   = lane & 15;
    const int dbase = wave * 16;
    const int rowA  = dbase + l16;
    const int offA  = half * 8;
    const int offB  = half * 8;

    // Resident A fragments for Wk^T and Wv^T (K = 128 -> 4 k-steps).
    v16bf ak[4], av[4];
#pragma unroll
    for (int ks = 0; ks < 4; ++ks) {
        ak[ks] = load_fragA(&WtK[rowA * LDST + ks * 32 + offA]);
        av[ks] = load_fragA(&WtV[rowA * LDST + ks * 32 + offA]);
    }

    float bkv[8], bvv[8];
#pragma unroll
    for (int r = 0; r < 8; ++r) {
        const int d = dbase + r + half * 8;
        bkv[r] = bk[d];
        bvv[r] = bv[d];
    }

    float accS[8], accP[8];
#pragma unroll
    for (int r = 0; r < 8; ++r) { accS[r] = 0.0f; accP[r] = 0.0f; }

    // Base LDS pointer for this lane's B column; step a whole l-tile at a time.
    const __bf16* bbase = &xT[l16 * LDST + offB];
    const int tileStep = 16 * LDST;

    v16bf b0[4], b1[4];
    load_fragB4(b0, bbase);

#pragma unroll
    for (int lt = 0; lt < LTILES; lt += 2) {
        // ---- tile lt (buffer b0); prefetch lt+1 into b1 first ----
        load_fragB4(b1, bbase + (lt + 1) * tileStep);
        {
            v8f kt = {};
            v8f vt = {};
#pragma unroll
            for (int ks = 0; ks < 4; ++ks) kt = wmma_bf16(ak[ks], b0[ks], kt);
#pragma unroll
            for (int ks = 0; ks < 4; ++ks) vt = wmma_bf16(av[ks], b0[ks], vt);
#pragma unroll
            for (int r = 0; r < 8; ++r) {
                const float e = __expf(kt[r] + bkv[r]);
                accS[r] += e;
                accP[r] += e * (vt[r] + bvv[r]);
            }
        }
        // ---- tile lt+1 (buffer b1); prefetch lt+2 into b0 first ----
        if (lt + 2 < LTILES) load_fragB4(b0, bbase + (lt + 2) * tileStep);
        {
            v8f kt = {};
            v8f vt = {};
#pragma unroll
            for (int ks = 0; ks < 4; ++ks) kt = wmma_bf16(ak[ks], b1[ks], kt);
#pragma unroll
            for (int ks = 0; ks < 4; ++ks) vt = wmma_bf16(av[ks], b1[ks], vt);
#pragma unroll
            for (int r = 0; r < 8; ++r) {
                const float e = __expf(kt[r] + bkv[r]);
                accS[r] += e;
                accP[r] += e * (vt[r] + bvv[r]);
            }
        }
    }

    // Butterfly-reduce the 16 lanes that share each d (lane halves independent).
#pragma unroll
    for (int r = 0; r < 8; ++r) {
#pragma unroll
        for (int m = 1; m <= 8; m <<= 1) {
            accS[r] += __shfl_xor(accS[r], m, 32);
            accP[r] += __shfl_xor(accP[r], m, 32);
        }
    }
    if (l16 == 0) {
#pragma unroll
        for (int r = 0; r < 8; ++r) {
            const int d = dbase + r + half * 8;
            atomicAdd(&wsDen[n * DIM + d], accS[r]);
            atomicAdd(&wsNum[n * DIM + d], accP[r]);
        }
    }
}

// ---------------------------------------------------------------------------
// Kernel 3: Q projection + sigmoid * pooled, written straight to out[n,d,l].
// ---------------------------------------------------------------------------
__global__ __launch_bounds__(256) void aft_out_kernel(
    const float* __restrict__ x,
    const float* __restrict__ Wq, const float* __restrict__ bq,
    const float* __restrict__ wsNum, const float* __restrict__ wsDen,
    float* __restrict__ out)
{
    extern __shared__ __align__(16) char smem[];
    __bf16* xT  = (__bf16*)(smem);                          // LT  x LDST
    __bf16* WtQ = (__bf16*)(smem + (size_t)LT * LDST * 2);  // DIM x LDST

    const int tid = threadIdx.x;
    const int n   = blockIdx.y;
    const int l0  = blockIdx.x * LT;

    stage_weightT(Wq, WtQ, tid);
    stage_xT(x + (size_t)n * DIM * SEQ + l0, xT, tid);
    __syncthreads();

    const int wave  = tid >> 5;
    const int lane  = tid & 31;
    const int half  = lane >> 4;
    const int l16   = lane & 15;
    const int dbase = wave * 16;
    const int rowA  = dbase + l16;
    const int offA  = half * 8;
    const int offB  = half * 8;

    v16bf aq[4];
#pragma unroll
    for (int ks = 0; ks < 4; ++ks)
        aq[ks] = load_fragA(&WtQ[rowA * LDST + ks * 32 + offA]);

    float pool[8], bqv[8];
#pragma unroll
    for (int r = 0; r < 8; ++r) {
        const int d = dbase + r + half * 8;
        pool[r] = wsNum[n * DIM + d] / wsDen[n * DIM + d];
        bqv[r]  = bq[d];
    }

    const __bf16* bbase = &xT[l16 * LDST + offB];
    const int tileStep = 16 * LDST;

    v16bf b0[4], b1[4];
    load_fragB4(b0, bbase);

#pragma unroll
    for (int lt = 0; lt < LTILES; lt += 2) {
        load_fragB4(b1, bbase + (lt + 1) * tileStep);
        {
            v8f qt = {};
#pragma unroll
            for (int ks = 0; ks < 4; ++ks) qt = wmma_bf16(aq[ks], b0[ks], qt);
#pragma unroll
            for (int r = 0; r < 8; ++r) {
                const int d = dbase + r + half * 8;
                const int l = l0 + lt * 16 + l16;
                const float s = 1.0f / (1.0f + __expf(-(qt[r] + bqv[r])));
                out[(size_t)(n * DIM + d) * SEQ + l] = s * pool[r];
            }
        }
        if (lt + 2 < LTILES) load_fragB4(b0, bbase + (lt + 2) * tileStep);
        {
            v8f qt = {};
#pragma unroll
            for (int ks = 0; ks < 4; ++ks) qt = wmma_bf16(aq[ks], b1[ks], qt);
#pragma unroll
            for (int r = 0; r < 8; ++r) {
                const int d = dbase + r + half * 8;
                const int l = l0 + (lt + 1) * 16 + l16;
                const float s = 1.0f / (1.0f + __expf(-(qt[r] + bqv[r])));
                out[(size_t)(n * DIM + d) * SEQ + l] = s * pool[r];
            }
        }
    }
}

// ---------------------------------------------------------------------------
extern "C" void kernel_launch(void* const* d_in, const int* in_sizes, int n_in,
                              void* d_out, int out_size, void* d_ws, size_t ws_size,
                              hipStream_t stream) {
    const float* x  = (const float*)d_in[0];
    const float* Wq = (const float*)d_in[1];
    const float* bq = (const float*)d_in[2];
    const float* Wk = (const float*)d_in[3];
    const float* bk = (const float*)d_in[4];
    const float* Wv = (const float*)d_in[5];
    const float* bv = (const float*)d_in[6];
    float* out = (float*)d_out;

    float* wsNum = (float*)d_ws;           // [NB*DIM]
    float* wsDen = wsNum + NB * DIM;       // [NB*DIM]

    const int wsCount = 2 * NB * DIM;
    aft_zero_kernel<<<(wsCount + 255) / 256, 256, 0, stream>>>(wsNum, wsCount);

    dim3 grid(SEQ / LT, NB);
    const size_t smemPool = (size_t)(LT + 2 * DIM) * LDST * 2;  // 139,264 B
    const size_t smemOut  = (size_t)(LT + DIM) * LDST * 2;      // 104,448 B

    aft_pool_kernel<<<grid, 256, smemPool, stream>>>(x, Wk, bk, Wv, bv, wsNum, wsDen);
    aft_out_kernel<<<grid, 256, smemOut, stream>>>(x, Wq, bq, wsNum, wsDen, out);
}
